// Attention_org_37177236914609
// MI455X (gfx1250) — matrix operand order repaired and checked by reading, not compiled, not run on hardware
//
#include <hip/hip_runtime.h>
#include <hip/hip_bf16.h>
#include <math.h>

// ---------------------------------------------------------------------------
// Attention_org on MI455X (gfx1250, wave32, WMMA).
// bf16 NT GEMMs on v_wmma_f32_16x16x32_bf16, double-buffered LDS staging via
// GLOBAL_LOAD_ASYNC_TO_LDS_B128 (ASYNCcnt) when the builtin is available.
// ---------------------------------------------------------------------------

typedef unsigned short ushort_t;
typedef __attribute__((ext_vector_type(16))) __bf16 bfx16;
typedef __attribute__((ext_vector_type(8)))  __bf16 bfx8;
typedef __attribute__((ext_vector_type(8)))  float  fx8;

#define B_   4
#define N_   1024
#define KV_  1984
#define H_   4

#define BM 128
#define BN 128
#define BK 32
#define LDS_STRIDE 40   // 32 + 8 halfs pad; 80 bytes keeps 16B alignment

#if defined(__HIP_DEVICE_COMPILE__) && __has_builtin(__builtin_amdgcn_global_load_async_to_lds_b128)
#define ATT_ASYNC_LDS 1
// exact pointer types per the compiler diagnostic: v4i addrspace(1)* source
typedef __attribute__((__vector_size__(4 * sizeof(int)))) int att_v4i;
typedef __attribute__((address_space(1))) att_v4i* att_gp;
typedef __attribute__((address_space(3))) att_v4i* att_lp;
#else
#define ATT_ASYNC_LDS 0
#endif

#if defined(__HIP_DEVICE_COMPILE__) && __has_builtin(__builtin_amdgcn_s_wait_asynccnt)
#define ATT_WAIT_ASYNC() __builtin_amdgcn_s_wait_asynccnt(0)
#elif defined(__HIP_DEVICE_COMPILE__)
#define ATT_WAIT_ASYNC() asm volatile("s_wait_asynccnt 0x0" ::: "memory")
#else
#define ATT_WAIT_ASYNC() ((void)0)
#endif

__device__ __forceinline__ ushort_t f2bf(float f) {
  unsigned u = __builtin_bit_cast(unsigned, f);
  u += 0x7FFFu + ((u >> 16) & 1u);      // round-to-nearest-even
  return (ushort_t)(u >> 16);
}

// ------------------------------ conversion --------------------------------
__global__ __launch_bounds__(256)
void cvt_f32_bf16(const float* __restrict__ x, ushort_t* __restrict__ y, long n) {
  long i = (long)blockIdx.x * 256 + threadIdx.x;
  long stride = (long)gridDim.x * 256;
  for (; i < n; i += stride) y[i] = f2bf(x[i]);
}

// ------------------------------ NT GEMM ------------------------------------
// C[z][m][n] = alpha * sum_{kb} sum_k A[z,kb][m,k] * B[z,kb][n,k]
// z = bi*nh + hi with independent (b,h) strides for A/B; kb batches add an
// extra accumulation dimension (used for the over-heads context sum).
template<bool OBF>
__global__ __launch_bounds__(256)
void gemm_nt(const ushort_t* __restrict__ A, const ushort_t* __restrict__ Bm,
             void* __restrict__ Cv,
             int M, int Nn, int Kd, int lda, int ldb, int ldc,
             int nh, long sAb, long sAh, long sBb, long sBh, long sCz,
             int kb, long sAk, long sBk, float alpha)
{
  __shared__ __align__(16) ushort_t As[2][BM * LDS_STRIDE];
  __shared__ __align__(16) ushort_t Bs[2][BN * LDS_STRIDE];

  const int tid  = threadIdx.x;
  const int lane = tid & 31;
  const int wid  = tid >> 5;
  const int wm   = (wid & 3) * 32;   // wave row offset (4 waves in M)
  const int wn   = (wid >> 2) * 64;  // wave col offset (2 waves in N)
  const int m0   = blockIdx.x * BM;
  const int n0   = blockIdx.y * BN;
  const int z    = blockIdx.z;
  const int bi   = z / nh, hi = z % nh;
  const ushort_t* Ab0 = A  + (long)bi * sAb + (long)hi * sAh;
  const ushort_t* Bb0 = Bm + (long)bi * sBb + (long)hi * sBh;

  const int hr   = lane & 15;   // row/col within 16
  const int hsel = lane >> 4;   // half-wave select

  fx8 acc[2][4];
  #pragma unroll
  for (int mf = 0; mf < 2; ++mf)
    #pragma unroll
    for (int nf = 0; nf < 4; ++nf)
      #pragma unroll
      for (int e = 0; e < 8; ++e) acc[mf][nf][e] = 0.0f;

  // Stage one 128x32 tile pair into LDS buffer `sel`. Async path writes
  // L2 -> LDS directly (ASYNCcnt); OOB rows are zero-filled with ds_store.
  auto stage = [&](int sel, const ushort_t* Ab, const ushort_t* Bb, int k0) {
    #pragma unroll
    for (int rep = 0; rep < 2; ++rep) {
      int id  = tid + rep * 256;      // 512 chunks of 8 halfs
      int row = id >> 2;
      int ch  = (id & 3) * 8;
      int gm  = m0 + row;
      int gn  = n0 + row;
      ushort_t* la = &As[sel][row * LDS_STRIDE + ch];
      ushort_t* lb = &Bs[sel][row * LDS_STRIDE + ch];
#if ATT_ASYNC_LDS
      if (gm < M) {
        __builtin_amdgcn_global_load_async_to_lds_b128(
            (att_gp)(Ab + (long)gm * lda + k0 + ch), (att_lp)la, 0, 0);
      } else {
        uint4 zz{0u, 0u, 0u, 0u}; *(uint4*)la = zz;
      }
      if (gn < Nn) {
        __builtin_amdgcn_global_load_async_to_lds_b128(
            (att_gp)(Bb + (long)gn * ldb + k0 + ch), (att_lp)lb, 0, 0);
      } else {
        uint4 zz{0u, 0u, 0u, 0u}; *(uint4*)lb = zz;
      }
#else
      uint4 va{0u, 0u, 0u, 0u};
      if (gm < M) {
        va = *(const uint4*)(Ab + (long)gm * lda + k0 + ch);
        __builtin_prefetch(Ab + (long)gm * lda + k0 + ch + BK, 0, 1);
      }
      *(uint4*)la = va;
      uint4 vb{0u, 0u, 0u, 0u};
      if (gn < Nn) {
        vb = *(const uint4*)(Bb + (long)gn * ldb + k0 + ch);
        __builtin_prefetch(Bb + (long)gn * ldb + k0 + ch + BK, 0, 1);
      }
      *(uint4*)lb = vb;
#endif
    }
  };

  const int kSteps = Kd / BK;          // all K dims are multiples of 32
  const int total  = kb * kSteps;

  stage(0, Ab0, Bb0, 0);               // prologue

  for (int s = 0; s < total; ++s) {
    const int sel = s & 1;
    ATT_WAIT_ASYNC();                  // loads into buffer `sel` complete
    __syncthreads();                   // visible to all waves; frees sel^1

    if (s + 1 < total) {               // stage next tile into other buffer
      int kb2 = (s + 1) / kSteps;
      int kk2 = (s + 1) % kSteps;
      stage(sel ^ 1, Ab0 + (long)kb2 * sAk, Bb0 + (long)kb2 * sBk, kk2 * BK);
    }

    // Build WMMA fragments per documented 16-bit A/B VGPR layouts.
    bfx16 af[2], bfm[4];
    #pragma unroll
    for (int mf = 0; mf < 2; ++mf) {
      int r = wm + mf * 16 + hr;
      int kbase = hsel * 8;   // lanes 0-15: K 0..7/16..23, lanes 16-31: 8..15/24..31
      bfx8 lo = *reinterpret_cast<const bfx8*>(&As[sel][r * LDS_STRIDE + kbase]);
      bfx8 hv = *reinterpret_cast<const bfx8*>(&As[sel][r * LDS_STRIDE + kbase + 16]);
      #pragma unroll
      for (int e = 0; e < 8; ++e) { af[mf][e] = lo[e]; af[mf][e + 8] = hv[e]; }
    }
    #pragma unroll
    for (int nf = 0; nf < 4; ++nf) {
      int cidx = wn + nf * 16 + hr;
      int klo  = hsel * 8;    // lanes 0-15: K 0..7/16..23, lanes 16-31: 8..15/24..31
      bfx8 lo = *reinterpret_cast<const bfx8*>(&Bs[sel][cidx * LDS_STRIDE + klo]);
      bfx8 hv = *reinterpret_cast<const bfx8*>(&Bs[sel][cidx * LDS_STRIDE + klo + 16]);
      #pragma unroll
      for (int e = 0; e < 8; ++e) { bfm[nf][e] = lo[e]; bfm[nf][e + 8] = hv[e]; }
    }

    #pragma unroll
    for (int mf = 0; mf < 2; ++mf)
      #pragma unroll
      for (int nf = 0; nf < 4; ++nf)
        acc[mf][nf] = __builtin_amdgcn_wmma_f32_16x16x32_bf16(
            false, af[mf], false, bfm[nf], (short)0, acc[mf][nf], false, false);
  }

  // Epilogue: C/D layout (VGPR i -> M=i / M=8+i, N=lane&15).
  const long cbase = (long)z * sCz;
  #pragma unroll
  for (int mf = 0; mf < 2; ++mf)
    #pragma unroll
    for (int nf = 0; nf < 4; ++nf)
      #pragma unroll
      for (int i = 0; i < 8; ++i) {
        int gm = m0 + wm + mf * 16 + i + hsel * 8;
        int gn = n0 + wn + nf * 16 + hr;
        if (gm < M && gn < Nn) {
          float v = alpha * acc[mf][nf][i];
          if (OBF) ((ushort_t*)Cv)[cbase + (long)gm * ldc + gn] = f2bf(v);
          else     ((float*)Cv)[cbase + (long)gm * ldc + gn] = v;
        }
      }
}

// -------------------- InstanceNorm stats (per b,h) -------------------------
__global__ __launch_bounds__(256)
void instnorm_stats(const float* __restrict__ S, float* __restrict__ stats, long per) {
  const int z = blockIdx.x;
  const float* p = S + (long)z * per;
  float s = 0.f, ss = 0.f;
  for (long i = threadIdx.x; i < per; i += 256) { float x = p[i]; s += x; ss += x * x; }
  __shared__ float r1[256], r2[256];
  const int tid = threadIdx.x;
  r1[tid] = s; r2[tid] = ss; __syncthreads();
  for (int off = 128; off > 0; off >>= 1) {
    if (tid < off) { r1[tid] += r1[tid + off]; r2[tid] += r2[tid + off]; }
    __syncthreads();
  }
  if (tid == 0) {
    float mu  = r1[0] / (float)per;
    float var = r2[0] / (float)per - mu * mu;
    stats[2 * z]     = mu;
    stats[2 * z + 1] = rsqrtf(fmaxf(var, 0.f) + 1e-5f);
  }
}

// ------------- normalize + row softmax over KV, emit bf16 probs ------------
__global__ __launch_bounds__(256)
void norm_softmax(const float* __restrict__ S, ushort_t* __restrict__ P,
                  const float* __restrict__ stats, int kvlen, int rows_per_bh) {
  const long r  = blockIdx.x;
  const int  bh = (int)(r / rows_per_bh);
  const float mu = stats[2 * bh], rstd = stats[2 * bh + 1];
  const float* srow = S + r * (long)kvlen;
  ushort_t*    prow = P + r * (long)kvlen;
  const int tid = threadIdx.x;
  float v[8];
  float mx = -3.0e38f;
  #pragma unroll
  for (int j = 0; j < 8; ++j) {
    int idx = tid + j * 256;
    v[j] = (idx < kvlen) ? (srow[idx] - mu) * rstd : -3.0e38f;
    mx = fmaxf(mx, v[j]);
  }
  __shared__ float red[256];
  red[tid] = mx; __syncthreads();
  for (int off = 128; off > 0; off >>= 1) {
    if (tid < off) red[tid] = fmaxf(red[tid], red[tid + off]);
    __syncthreads();
  }
  mx = red[0]; __syncthreads();
  float sum = 0.f;
  #pragma unroll
  for (int j = 0; j < 8; ++j) {
    int idx = tid + j * 256;
    if (idx < kvlen) { v[j] = __expf(v[j] - mx); sum += v[j]; }
  }
  red[tid] = sum; __syncthreads();
  for (int off = 128; off > 0; off >>= 1) {
    if (tid < off) red[tid] += red[tid + off];
    __syncthreads();
  }
  const float inv = 1.0f / red[0];
  #pragma unroll
  for (int j = 0; j < 8; ++j) {
    int idx = tid + j * 256;
    if (idx < kvlen) prow[idx] = f2bf(v[j] * inv);
  }
}

// ------------------------------- host side ---------------------------------
static void run_gemm(bool obf, const ushort_t* A, const ushort_t* Bm, void* C,
                     int M, int Nn, int Kd, int lda, int ldb, int ldc,
                     int nbz, int nh,
                     long sAb, long sAh, long sBb, long sBh, long sCz,
                     int kb, long sAk, long sBk, float alpha, hipStream_t st) {
  dim3 g((M + BM - 1) / BM, (Nn + BN - 1) / BN, nbz);
  dim3 b(256);
  if (obf)
    gemm_nt<true><<<g, b, 0, st>>>(A, Bm, C, M, Nn, Kd, lda, ldb, ldc,
                                   nh, sAb, sAh, sBb, sBh, sCz, kb, sAk, sBk, alpha);
  else
    gemm_nt<false><<<g, b, 0, st>>>(A, Bm, C, M, Nn, Kd, lda, ldb, ldc,
                                    nh, sAb, sAh, sBb, sBh, sCz, kb, sAk, sBk, alpha);
}

static void run_cvt(const float* x, ushort_t* y, long n, hipStream_t st) {
  long blocks = (n + 255) / 256;
  if (blocks > 8192) blocks = 8192;
  cvt_f32_bf16<<<dim3((unsigned)blocks), dim3(256), 0, st>>>(x, y, n);
}

extern "C" void kernel_launch(void* const* d_in, const int* in_sizes, int n_in,
                              void* d_out, int out_size, void* d_ws, size_t ws_size,
                              hipStream_t stream) {
  (void)in_sizes; (void)n_in; (void)out_size; (void)ws_size;
  const int  Cs[5]     = {64, 128, 256, 512, 1024};
  const long outOff[5] = {0, 262144, 786432, 1835008, 3932160};
  const int  Cmax      = 1024;
  const float inv_sqrt_kv = 1.0f / sqrtf((float)KV_);

  // inputs (setup_inputs order)
  const float* emb[5] = {(const float*)d_in[0], (const float*)d_in[1], (const float*)d_in[2],
                         (const float*)d_in[3], (const float*)d_in[4]};
  const float* emb_all = (const float*)d_in[5];
  const float* qw[5]   = {(const float*)d_in[6], (const float*)d_in[7], (const float*)d_in[8],
                          (const float*)d_in[9], (const float*)d_in[10]};
  const float* kw = (const float*)d_in[11];
  const float* vw = (const float*)d_in[12];
  const float* ow[5] = {(const float*)d_in[13], (const float*)d_in[14], (const float*)d_in[15],
                        (const float*)d_in[16], (const float*)d_in[17]};

  // workspace carve-up (256B aligned)
  char* ws = (char*)d_ws;
  size_t o = 0;
  auto carve = [&](size_t bytes) -> char* {
    char* p = ws + o; o = (o + bytes + 255) & ~(size_t)255; return p;
  };
  ushort_t* Ea    = (ushort_t*)carve((size_t)B_ * N_ * KV_ * 2);
  ushort_t* Kw16  = (ushort_t*)carve((size_t)H_ * KV_ * KV_ * 2);
  ushort_t* Vw16  = (ushort_t*)carve((size_t)H_ * KV_ * KV_ * 2);
  ushort_t* Kt    = (ushort_t*)carve((size_t)B_ * H_ * KV_ * N_ * 2);   // [B,H,KV,N]
  ushort_t* Vb    = (ushort_t*)carve((size_t)B_ * H_ * N_ * KV_ * 2);   // [B,H,N,KV]
  ushort_t* E16   = (ushort_t*)carve((size_t)B_ * N_ * Cmax * 2);
  ushort_t* Qw16  = (ushort_t*)carve((size_t)H_ * Cmax * Cmax * 2);
  ushort_t* Ow16  = (ushort_t*)carve((size_t)Cmax * Cmax * 2);
  ushort_t* Qt    = (ushort_t*)carve((size_t)B_ * H_ * Cmax * N_ * 2);  // [B,H,C,N]
  float*    Sb    = (float*)carve((size_t)B_ * H_ * Cmax * KV_ * 4);    // [B,H,C,KV]
  ushort_t* Pb    = (ushort_t*)carve((size_t)B_ * H_ * Cmax * KV_ * 2);
  ushort_t* Ctx   = (ushort_t*)carve((size_t)B_ * N_ * Cmax * 2);       // [B,N,C]
  float*    stats = (float*)carve((size_t)B_ * H_ * 2 * sizeof(float));

  // 1) shared conversions
  run_cvt(emb_all, Ea,   (long)B_ * N_ * KV_,  stream);
  run_cvt(kw,      Kw16, (long)H_ * KV_ * KV_, stream);
  run_cvt(vw,      Vw16, (long)H_ * KV_ * KV_, stream);

  // 2) Kt[b,h,kv,n] = kw[h] x emb_all[b]^T     (M=KV, N=N, K=KV)
  run_gemm(true, Kw16, Ea, Kt, KV_, N_, KV_, KV_, KV_, N_,
           B_ * H_, H_, 0, (long)KV_ * KV_, (long)N_ * KV_, 0, (long)KV_ * N_,
           1, 0, 0, 1.0f, stream);

  // 3) V[b,h,n,kv] = emb_all[b] x vw[h]^T      (M=N, N=KV, K=KV)
  run_gemm(true, Ea, Vw16, Vb, N_, KV_, KV_, KV_, KV_, KV_,
           B_ * H_, H_, (long)N_ * KV_, 0, 0, (long)KV_ * KV_, (long)N_ * KV_,
           1, 0, 0, 1.0f, stream);

  // 4) branches
  for (int i = 0; i < 5; ++i) {
    const int c = Cs[i];
    run_cvt(emb[i], E16,  (long)B_ * N_ * c, stream);
    run_cvt(qw[i],  Qw16, (long)H_ * c * c,  stream);
    run_cvt(ow[i],  Ow16, (long)c * c,       stream);

    // Qt[b,h,c,n] = qw[h] x emb[b]^T          (M=c, N=N, K=c)
    run_gemm(true, Qw16, E16, Qt, c, N_, c, c, c, N_,
             B_ * H_, H_, 0, (long)c * c, (long)N_ * c, 0, (long)c * N_,
             1, 0, 0, 1.0f, stream);

    // s[b,h,c,kv] = (Qt x Kt^T) * inv_sqrt_kv (M=c, N=KV, K=N) -> f32
    run_gemm(false, Qt, Kt, Sb, c, KV_, N_, N_, N_, KV_,
             B_ * H_, H_, (long)H_ * c * N_, (long)c * N_,
             (long)H_ * KV_ * N_, (long)KV_ * N_, (long)c * KV_,
             1, 0, 0, inv_sqrt_kv, stream);

    // InstanceNorm stats + softmax -> P bf16
    instnorm_stats<<<dim3(B_ * H_), dim3(256), 0, stream>>>(Sb, stats, (long)c * KV_);
    norm_softmax<<<dim3(B_ * H_ * c), dim3(256), 0, stream>>>(Sb, Pb, stats, KV_, c);

    // ctx[b,n,c] = (1/H) * sum_h V[b,h] x P[b,h]^T  (M=N, N=c, K=KV, kb=H)
    run_gemm(true, Vb, Pb, Ctx, N_, c, KV_, KV_, KV_, c,
             B_, 1, (long)H_ * N_ * KV_, 0, (long)H_ * c * KV_, 0, (long)N_ * c,
             H_, (long)N_ * KV_, (long)c * KV_, 1.0f / (float)H_, stream);

    // out[b,n,o] = ctx[b] x ow^T               (M=N, N=c, K=c) -> f32 d_out
    run_gemm(false, Ctx, Ow16, (float*)d_out + outOff[i], N_, c, c, c, c, c,
             B_, 1, (long)N_ * c, 0, 0, 0, (long)N_ * c,
             1, 0, 0, 1.0f, stream);
  }
}
